// Seq2Seq_49624052138142
// MI455X (gfx1250) — matrix operand order repaired and checked by reading, not compiled, not run on hardware
//
#include <hip/hip_runtime.h>

// ---------------------------------------------------------------------------
// Seq2Seq LSTM (B=32, TS=128, TT=64, E=256, H=512, V=30000) for gfx1250.
//  - bf16 WMMA (v_wmma_f32_16x16x32_bf16) everywhere, fp32 accumulate
//  - ONE persistent kernel for encoder+decoder: gate-interleaved weight
//    packing makes the LSTM cell wave-local (__shfl on wave32), cell state c
//    lives in registers for all 191 steps, h published as packed bf16
//    fragments through a ping-pong buffer -> 1 grid barrier per timestep
//  - branch-free sigmoid/tanh (exp-based) keeps the recurrent critical path
//    free of saveexec divergence
//  - output projection: LDS-staged A fragments, 5 N-tiles per wave,
//    K-loop kept rolled (unroll 1) to avoid the register-spill pathology
// ---------------------------------------------------------------------------

typedef __attribute__((ext_vector_type(16))) __bf16 v16bf;
typedef __attribute__((ext_vector_type(8)))  float  v8f;

#define NWG_LSTM 16
#define NT_FC    1875              // 30000 / 16
#define MT_FC    126               // 2016 / 16
#define NTW      5                 // N-tiles per wave in projection
#define NGRP_FC  47                // ceil(1875 / (8 waves * 5))

// ---- workspace layout (bytes) ---------------------------------------------
constexpr size_t OFF_W0E = 0;
constexpr size_t OFF_W1E = OFF_W0E + 2048ull * 768 * 2;
constexpr size_t OFF_W0D = OFF_W1E + 2048ull * 1024 * 2;
constexpr size_t OFF_W1D = OFF_W0D + 2048ull * 768 * 2;
constexpr size_t OFF_WFC = OFF_W1D + 2048ull * 1024 * 2;
constexpr size_t OFF_A0E = OFF_WFC + 30000ull * 512 * 2;
constexpr size_t OFF_A0D = OFF_A0E + 128ull * 32 * 768 * 2;
constexpr size_t OFF_A1  = OFF_A0D + 63ull * 32 * 768 * 2;   // 2 x 64KB ping-pong
constexpr size_t OFF_HSA = OFF_A1  + 2ull * 32 * 1024 * 2;
constexpr size_t OFF_BAR = OFF_HSA + 2016ull * 512 * 2;

// ---- fragment index helpers (element index, bf16 units) -------------------
// A matrix 16x32 bf16 (MxK): lane l (m=l%16, half=l/16) holds
//   VGPR j<4 : K = half*8 + 2j + s ;  j>=4 : K = 16 + half*8 + 2(j-4) + s
__device__ __forceinline__ size_t apos(int KT, int m, int k) {
    int mt = m >> 4, ml = m & 15;
    int kt = k >> 5, kk = k & 31;
    int half = (kk & 15) >> 3;
    int lane = half * 16 + ml;
    int blk  = kk >> 4;
    int j    = blk * 4 + ((kk & 7) >> 1);
    int s    = kk & 1;
    return (((size_t)(mt * KT + kt) * 32 + lane) * 16) + j * 2 + s;
}

// B matrix 32x16 bf16 (KxN): lane half h, element e holds
// K = 8h + e (e<8), K = 16 + 8h + (e-8) (e>=8); N = lane % 16.
__device__ __forceinline__ size_t bpos(int KT, int n, int k) {
    int nt = n >> 4, nl = n & 15;
    int kt = k >> 5, kk = k & 31;
    int lane = ((kk >> 3) & 1) * 16 + nl;
    int j    = (kk >> 4) * 4 + ((kk & 7) >> 1);
    int s    = kk & 1;
    return (((size_t)(nt * KT + kt) * 32 + lane) * 16) + j * 2 + s;
}

// branch-free activations (single v_exp_f32 each, no EXEC manipulation)
__device__ __forceinline__ float sigmoidf(float x) {
    return 1.0f / (1.0f + __expf(-x));
}
__device__ __forceinline__ float tanh_fast(float x) {
    return 2.0f / (1.0f + __expf(-2.0f * x)) - 1.0f;
}

// ---------------------------------------------------------------------------
// weight / embedding packing
// ---------------------------------------------------------------------------
// Gate-interleaved column permutation: original row n = gate*512 + u
// -> packed column n' = (u/4)*16 + gate*4 + (u%4), so wave w = ntile owns
// all four gates of units 4w..4w+3.
__global__ void pack_lstm_w(const float* __restrict__ Wih,
                            const float* __restrict__ Whh,
                            int Ein, int K, __bf16* __restrict__ dst) {
    size_t total  = 2048ull * K;
    size_t stride = (size_t)gridDim.x * blockDim.x;
    for (size_t i = (size_t)blockIdx.x * blockDim.x + threadIdx.x; i < total; i += stride) {
        int n = (int)(i / K), k = (int)(i % K);
        float v = (k < Ein) ? Wih[(size_t)n * Ein + k] : Whh[(size_t)n * 512 + (k - Ein)];
        int gate = n >> 9, u = n & 511;
        int np   = ((u >> 2) << 4) | (gate << 2) | (u & 3);
        dst[bpos(K >> 5, np, k)] = (__bf16)v;
    }
}

__global__ void pack_fc_w(const float* __restrict__ W, __bf16* __restrict__ dst) {
    size_t total  = 30000ull * 512;
    size_t stride = (size_t)gridDim.x * blockDim.x;
    for (size_t i = (size_t)blockIdx.x * blockDim.x + threadIdx.x; i < total; i += stride) {
        int n = (int)(i / 512), k = (int)(i & 511);
        dst[bpos(16, n, k)] = (__bf16)W[i];
    }
}

// gather embedding rows and write straight into packed A-fragment layout
__global__ void embed_pack(const int* __restrict__ tok, int T, int tokStride,
                           const float* __restrict__ emb,
                           __bf16* __restrict__ A0) {
    size_t total  = (size_t)T * 32 * 256;
    size_t stride = (size_t)gridDim.x * blockDim.x;
    for (size_t i = (size_t)blockIdx.x * blockDim.x + threadIdx.x; i < total; i += stride) {
        int t = (int)(i / (32 * 256));
        int rem = (int)(i % (32 * 256));
        int b = rem >> 8, k = rem & 255;
        int row = tok[b * tokStride + t];
        float v = emb[(size_t)row * 256 + k];
        A0[(size_t)t * 24576 + apos(24, b, k)] = (__bf16)v;
    }
}

__global__ void init_ws(__bf16* A1, __bf16* A0enc, unsigned* bar) {
    int i = blockIdx.x * 256 + threadIdx.x;      // 0..16383
#pragma unroll
    for (int r = 0; r < 4; ++r) A1[i + r * 16384] = (__bf16)0.f;   // both buffers
    int m = i >> 9, u = i & 511;
    A0enc[apos(24, m, 256 + u)] = (__bf16)0.f;   // zero initial h in A0[t=0]
    if (i == 0) { bar[0] = 0u; bar[1] = 0u; }
}

// ---------------------------------------------------------------------------
// persistent fused encoder+decoder LSTM
// ---------------------------------------------------------------------------
__device__ __forceinline__ void gridbar(unsigned* bar) {
    __syncthreads();
    if (threadIdx.x == 0) {
        __threadfence();
        volatile unsigned* gen = bar + 1;
        unsigned g = *gen;
        if (atomicAdd(bar, 1u) == gridDim.x - 1u) {
            bar[0] = 0u;
            __threadfence();
            *gen = g + 1u;
        } else {
            while (*gen == g) __builtin_amdgcn_s_sleep(2);
        }
    }
    __syncthreads();
    __threadfence();
}

// one wave: one 16-wide (gate-interleaved) N stripe over both M tiles
__device__ __forceinline__ void gemm2(const v16bf* __restrict__ A,
                                      const v16bf* __restrict__ W,
                                      int KT, const float* __restrict__ bias,
                                      int wave, int lane, v8f& acc0, v8f& acc1) {
    // column n' = wave*16 + (lane&15) decodes to gate=(lane&15)>>2, u=wave*4+(lane&3)
    float bv = bias[(((lane & 15) >> 2) << 9) + (wave << 2) + (lane & 3)];
    acc0 = (v8f){bv, bv, bv, bv, bv, bv, bv, bv};
    acc1 = acc0;
#pragma unroll 4
    for (int kt = 0; kt < KT; ++kt) {
        v16bf bf = W[((size_t)wave * KT + kt) * 32 + lane];
        v16bf a0 = A[(size_t)kt * 32 + lane];
        v16bf a1 = A[((size_t)KT + kt) * 32 + lane];
        acc0 = __builtin_amdgcn_wmma_f32_16x16x32_bf16(false, a0, false, bf,
                                                       (short)0, acc0, false, false);
        acc1 = __builtin_amdgcn_wmma_f32_16x16x32_bf16(false, a1, false, bf,
                                                       (short)0, acc1, false, false);
    }
}

// wave-local LSTM cell: gather i,f,g,o across lanes of the same unit group
__device__ __forceinline__ void cell8(const v8f& acc, float* cr, float* hv, int basel) {
#pragma unroll
    for (int j = 0; j < 8; ++j) {
        float iv = __shfl(acc[j], basel,      32);
        float fv = __shfl(acc[j], basel + 4,  32);
        float gv = __shfl(acc[j], basel + 8,  32);
        float ov = __shfl(acc[j], basel + 12, 32);
        float cc = sigmoidf(fv) * cr[j] + sigmoidf(iv) * tanh_fast(gv);
        cr[j] = cc;
        hv[j] = sigmoidf(ov) * tanh_fast(cc);
    }
}

__device__ __forceinline__ void lstm_phase(
    __bf16* __restrict__ A0, int T,
    const v16bf* __restrict__ W0, const float* __restrict__ b0,
    const v16bf* __restrict__ W1, const float* __restrict__ b1,
    __bf16* __restrict__ A1,                 // 2 x 32768 bf16 ping-pong
    __bf16* __restrict__ hsA,                // decoder only (else null)
    __bf16* __restrict__ nextA0,             // encoder only (else null)
    unsigned* bar, float (&c0r)[16], float (&c1r)[16], int wave, int lane) {

    const int  du     = lane & 3;
    const int  half   = lane >> 4;
    const int  basel  = (lane & 16) | du;
    const bool writer = (lane & 15) < 4;
    const int  u      = (wave << 2) | du;

    for (int t = 0; t < T; ++t) {
        const int p = t & 1;
        v8f acc0, acc1;
        float h0v[16], h1v[16];

        // ---- layer 0: gates = [x_t | h0] @ W0p + b0  (K = 768)
        gemm2((const v16bf*)A0 + (size_t)t * 1536, W0, 24, b0, wave, lane, acc0, acc1);
        cell8(acc0, &c0r[0], &h0v[0], basel);
        cell8(acc1, &c0r[8], &h0v[8], basel);
        if (writer) {
#pragma unroll
            for (int j = 0; j < 8; ++j) {
                int m0 = 8 * half + j, m1 = 16 + 8 * half + j;
                __bf16 hb0 = (__bf16)h0v[j], hb1 = (__bf16)h0v[8 + j];
                A1[p * 32768 + apos(32, m0, u)] = hb0;          // layer-1 input
                A1[p * 32768 + apos(32, m1, u)] = hb1;
                if (t + 1 < T) {
                    A0[(size_t)(t + 1) * 24576 + apos(24, m0, 256 + u)] = hb0;
                    A0[(size_t)(t + 1) * 24576 + apos(24, m1, 256 + u)] = hb1;
                } else if (nextA0) {
                    nextA0[apos(24, m0, 256 + u)] = hb0;        // seed decoder
                    nextA0[apos(24, m1, 256 + u)] = hb1;
                }
            }
        }
        gridbar(bar);   // the ONE barrier per step

        // ---- layer 1: gates = [h0 | h1] @ W1p + b1  (K = 1024)
        gemm2((const v16bf*)A1 + (size_t)p * 2048, W1, 32, b1, wave, lane, acc0, acc1);
        cell8(acc0, &c1r[0], &h1v[0], basel);
        cell8(acc1, &c1r[8], &h1v[8], basel);
        if (writer) {
#pragma unroll
            for (int j = 0; j < 8; ++j) {
                int m0 = 8 * half + j, m1 = 16 + 8 * half + j;
                __bf16 hb0 = (__bf16)h1v[j], hb1 = (__bf16)h1v[8 + j];
                A1[(p ^ 1) * 32768 + apos(32, m0, 512 + u)] = hb0;  // h1 for t+1
                A1[(p ^ 1) * 32768 + apos(32, m1, 512 + u)] = hb1;
                if (hsA) {
                    hsA[apos(16, t * 32 + m0, u)] = hb0;            // for projection
                    hsA[apos(16, t * 32 + m1, u)] = hb1;
                }
            }
        }
        // no trailing barrier needed: next step's post-cell0 barrier orders
        // cell1(t) -> GEMM1(t+1); GEMM0(t+1) touches only A0 (already ordered)
    }
}

__global__ __launch_bounds__(256)
void seq2seq_persistent(__bf16* __restrict__ A0e, __bf16* __restrict__ A0d,
                        const v16bf* __restrict__ W0e, const float* __restrict__ b0e,
                        const v16bf* __restrict__ W1e, const float* __restrict__ b1e,
                        const v16bf* __restrict__ W0d, const float* __restrict__ b0d,
                        const v16bf* __restrict__ W1d, const float* __restrict__ b1d,
                        __bf16* __restrict__ A1, __bf16* __restrict__ hsA,
                        unsigned* bar) {
    int wave = (blockIdx.x * 256 + threadIdx.x) >> 5;   // 0..127 <-> ntile
    int lane = threadIdx.x & 31;

    float c0r[16], c1r[16];
#pragma unroll
    for (int i = 0; i < 16; ++i) { c0r[i] = 0.f; c1r[i] = 0.f; }

    // encoder: 128 steps, seeds decoder A0[0] h-part; state stays in registers
    lstm_phase(A0e, 128, W0e, b0e, W1e, b1e, A1, (__bf16*)nullptr, A0d,
               bar, c0r, c1r, wave, lane);
    // decoder: 63 steps (128 even -> ping-pong parity continues correctly)
    lstm_phase(A0d, 63, W0d, b0d, W1d, b1d, A1, hsA, (__bf16*)nullptr,
               bar, c0r, c1r, wave, lane);
}

// ---------------------------------------------------------------------------
// output projection: (2016 x 512) @ (512 x 30000) + fc_b -> d_out
// ---------------------------------------------------------------------------
__global__ void zero_plane(float* __restrict__ out) {
    int i = blockIdx.x * 256 + threadIdx.x;      // 0..959999
    if (i < 32 * 30000) {
        int b = i / 30000, v = i % 30000;
        out[(size_t)b * 64 * 30000 + v] = 0.f;
    }
}

__global__ __launch_bounds__(256)
void proj_gemm(const v16bf* __restrict__ hsA, const v16bf* __restrict__ fcW,
               const float* __restrict__ fcb, float* __restrict__ out) {
    __shared__ v16bf Ash[16 * 32];               // 16 K-tiles of one M tile (16 KB)

    int mtile = blockIdx.x / NGRP_FC;
    int ngrp  = blockIdx.x % NGRP_FC;

    // cooperative stage of A fragments into LDS (shared by all 8 waves)
    {
        const uint4* s = (const uint4*)(hsA + (size_t)mtile * 512);
        uint4* d = (uint4*)Ash;
#pragma unroll
        for (int i = threadIdx.x; i < 1024; i += 256) d[i] = s[i];
    }
    __syncthreads();

    int wave = threadIdx.x >> 5;
    int lane = threadIdx.x & 31;
    int nt0  = ngrp * (8 * NTW) + wave * NTW;

    // clamp out-of-range tiles so the hot loop is branch-free (loads stay
    // in-bounds; only the stores are guarded)
    int ntc[NTW];
#pragma unroll
    for (int q = 0; q < NTW; ++q) {
        int nt = nt0 + q;
        ntc[q] = (nt < NT_FC) ? nt : (NT_FC - 1);
    }

    v8f acc[NTW];
#pragma unroll
    for (int q = 0; q < NTW; ++q) acc[q] = (v8f)0.f;

    // keep rolled: 1 ds_load (A) + NTW global B loads + NTW wmma per iter.
    // full unroll made the compiler preload all 16 A frags and spill to
    // scratch (observed in round-2 asm) -- unroll 1 keeps ~90 live VGPRs.
#pragma unroll 1
    for (int kt = 0; kt < 16; ++kt) {
        v16bf a = Ash[kt * 32 + lane];
        __builtin_prefetch(&fcW[((size_t)ntc[0] * 16 + kt + 1) * 32 + lane], 0, 1);
#pragma unroll
        for (int q = 0; q < NTW; ++q) {
            v16bf bf = fcW[((size_t)ntc[q] * 16 + kt) * 32 + lane];
            acc[q] = __builtin_amdgcn_wmma_f32_16x16x32_bf16(
                false, a, false, bf, (short)0, acc[q], false, false);
        }
    }

    int half = lane >> 4;
#pragma unroll
    for (int q = 0; q < NTW; ++q) {
        int nt = nt0 + q;
        if (nt < NT_FC) {
            int n = nt * 16 + (lane & 15);
            float bb = fcb[n];
#pragma unroll
            for (int j = 0; j < 8; ++j) {
                int m = mtile * 16 + 8 * half + j;   // global row = t*32 + b
                int t = m >> 5, b = m & 31;
                out[((size_t)b * 64 + t + 1) * 30000 + n] = acc[q][j] + bb;
            }
        }
    }
}

// ---------------------------------------------------------------------------
extern "C" void kernel_launch(void* const* d_in, const int* in_sizes, int n_in,
                              void* d_out, int out_size, void* d_ws, size_t ws_size,
                              hipStream_t stream) {
    const int*   src      = (const int*)d_in[0];
    const int*   trg      = (const int*)d_in[1];
    const float* enc_emb  = (const float*)d_in[2];
    const float* dec_emb  = (const float*)d_in[3];
    const float* enc_Wih0 = (const float*)d_in[4];
    const float* enc_Whh0 = (const float*)d_in[5];
    const float* enc_b0   = (const float*)d_in[6];
    const float* enc_Wih1 = (const float*)d_in[7];
    const float* enc_Whh1 = (const float*)d_in[8];
    const float* enc_b1   = (const float*)d_in[9];
    const float* dec_Wih0 = (const float*)d_in[10];
    const float* dec_Whh0 = (const float*)d_in[11];
    const float* dec_b0   = (const float*)d_in[12];
    const float* dec_Wih1 = (const float*)d_in[13];
    const float* dec_Whh1 = (const float*)d_in[14];
    const float* dec_b1   = (const float*)d_in[15];
    const float* fc_W     = (const float*)d_in[16];
    const float* fc_b     = (const float*)d_in[17];
    float* out = (float*)d_out;

    char* ws = (char*)d_ws;
    __bf16* W0e = (__bf16*)(ws + OFF_W0E);
    __bf16* W1e = (__bf16*)(ws + OFF_W1E);
    __bf16* W0d = (__bf16*)(ws + OFF_W0D);
    __bf16* W1d = (__bf16*)(ws + OFF_W1D);
    __bf16* Wfc = (__bf16*)(ws + OFF_WFC);
    __bf16* A0e = (__bf16*)(ws + OFF_A0E);
    __bf16* A0d = (__bf16*)(ws + OFF_A0D);
    __bf16* A1  = (__bf16*)(ws + OFF_A1);
    __bf16* hsA = (__bf16*)(ws + OFF_HSA);
    unsigned* bar = (unsigned*)(ws + OFF_BAR);

    // pack weights to bf16 WMMA fragment layouts (gate-interleaved for LSTM)
    pack_lstm_w<<<512, 256, 0, stream>>>(enc_Wih0, enc_Whh0, 256, 768, W0e);
    pack_lstm_w<<<512, 256, 0, stream>>>(enc_Wih1, enc_Whh1, 512, 1024, W1e);
    pack_lstm_w<<<512, 256, 0, stream>>>(dec_Wih0, dec_Whh0, 256, 768, W0d);
    pack_lstm_w<<<512, 256, 0, stream>>>(dec_Wih1, dec_Whh1, 512, 1024, W1d);
    pack_fc_w<<<2048, 256, 0, stream>>>(fc_W, Wfc);

    // embeddings -> packed A fragments (x part of layer-0 input, all steps)
    embed_pack<<<1024, 256, 0, stream>>>(src, 128, 128, enc_emb, A0e);
    embed_pack<<<1024, 256, 0, stream>>>(trg, 63, 64, dec_emb, A0d);

    // zero h seeds + ping-pong buffers + barrier state
    init_ws<<<64, 256, 0, stream>>>(A1, A0e, bar);

    // fused encoder+decoder, cell state register-resident, 1 barrier/step
    seq2seq_persistent<<<NWG_LSTM, 256, 0, stream>>>(
        A0e, A0d, (const v16bf*)W0e, enc_b0, (const v16bf*)W1e, enc_b1,
        (const v16bf*)W0d, dec_b0, (const v16bf*)W1d, dec_b1, A1, hsA, bar);

    // out[:, 0, :] = 0, then logits
    zero_plane<<<3750, 256, 0, stream>>>(out);
    proj_gemm<<<MT_FC * NGRP_FC, 256, 0, stream>>>(
        (const v16bf*)hsA, (const v16bf*)Wfc, fc_b, out);
}
